// BRITSAutoEncoder_61005715472502
// MI455X (gfx1250) — matrix operand order repaired and checked by reading, not compiled, not run on hardware
//
#include <hip/hip_runtime.h>

#define DEV __device__ __forceinline__

typedef __attribute__((ext_vector_type(16))) __bf16 bf16x16;
typedef __attribute__((ext_vector_type(8)))  __bf16 bf16x8;
typedef __attribute__((ext_vector_type(8)))  float  f32x8;

static constexpr int B = 128, T = 512, D = 64, H = 256, E = 64;

// ---- WMMA fragment loaders (CDNA5 16x16x32 bf16 layouts, wave32) ----
// A 16x32 (MxK): lane m=L&15, h=L>>4; per lane 16 bf16 at K = [k0+8h,+8) and [k0+16+8h,+8)
DEV bf16x16 frag_a(const __bf16* __restrict__ base, size_t ld, int lane, int k0) {
  const int m = lane & 15, h = lane >> 4;
  const __bf16* r = base + (size_t)m * ld + k0;
  bf16x8 lo = *(const bf16x8*)(r + 8 * h);
  bf16x8 hi = *(const bf16x8*)(r + 16 + 8 * h);
  bf16x16 a;
#pragma unroll
  for (int i = 0; i < 8; ++i) { a[i] = lo[i]; a[i + 8] = hi[i]; }
  return a;
}
// B 32x16 (KxN) with B[k][n] = W[n][k], W row-major (ld = K-stride):
// lane n=L&15, h=L>>4; 16 bf16 at W[n0+n][k0+8h,+8) and [k0+16+8h,+8)
DEV bf16x16 frag_b(const __bf16* __restrict__ w, size_t ld, int lane, int n0, int k0) {
  const int n = lane & 15, h = lane >> 4;
  const __bf16* r = w + (size_t)(n0 + n) * ld + k0;
  bf16x8 lo = *(const bf16x8*)(r + 8 * h);
  bf16x8 hi = *(const bf16x8*)(r + 16 + 8 * h);
  bf16x16 b;
#pragma unroll
  for (int i = 0; i < 8; ++i) { b[i] = lo[i]; b[i + 8] = hi[i]; }
  return b;
}
DEV f32x8 wmma(bf16x16 a, bf16x16 b, f32x8 c) {
  return __builtin_amdgcn_wmma_f32_16x16x32_bf16(false, a, false, b, (short)0, c,
                                                 false, false);
}
DEV float sigmoidf(float x) { return 1.0f / (1.0f + expf(-x)); }

// ================= prep kernels =================
__global__ void k_f2b(const float* __restrict__ s, __bf16* __restrict__ d, int n) {
  int i = blockIdx.x * blockDim.x + threadIdx.x;
  int stride = gridDim.x * blockDim.x;
  for (; i < n; i += stride) d[i] = (__bf16)s[i];
}
// extract Wih[:, 0:64] (x-part) as bf16, contiguous ld=64
__global__ void k_w1(const float* __restrict__ Wih, __bf16* __restrict__ d) {
  int i = blockIdx.x * blockDim.x + threadIdx.x;
  if (i < 3 * H * D) { int r = i / D, c = i % D; d[i] = (__bf16)Wih[(size_t)r * 3 * D + c]; }
}
// cg[j] = bih[j] + sum_c Wih[j, 64+c]   (m_t == 1 constant contribution; dn_t == 0)
__global__ void k_cg(const float* __restrict__ Wih, const float* __restrict__ bih,
                     float* __restrict__ cg) {
  int j = blockIdx.x * blockDim.x + threadIdx.x;
  if (j < 3 * H) {
    float s = bih[j];
    const float* r = Wih + (size_t)j * 3 * D + D;
    for (int c = 0; c < D; ++c) s += r[c];
    cg[j] = s;
  }
}
// gamma_h = exp(-relu(Wdh_b))  (Delta == 0 -> time-invariant)
__global__ void k_gamma(const float* __restrict__ b, float* __restrict__ g, int n) {
  int i = blockIdx.x * blockDim.x + threadIdx.x;
  if (i < n) g[i] = expf(-fmaxf(b[i], 0.0f));
}

// ================= encoder step (both directions) =================
// grid: (16 col-tiles of 16, 2 dirs), 256 thr = 8 waves; wave w owns batch rows [16w,16w+16)
__global__ void __launch_bounds__(256)
enc_step(const __bf16* __restrict__ Xb, const __bf16* __restrict__ W1b,
         const __bf16* __restrict__ Whhb, const float* __restrict__ cg,
         const float* __restrict__ gammah, const float* __restrict__ bhhf,
         const float* __restrict__ bhhb, const float* __restrict__ h_in,
         const __bf16* __restrict__ hs_in, float* __restrict__ h_out,
         __bf16* __restrict__ hs_out, float* __restrict__ hsum, int t) {
  const int dir = blockIdx.y;
  const int t_eff = dir ? (T - 1 - t) : t;
  const int j0 = blockIdx.x * 16;
  const int wave = threadIdx.x >> 5, lane = threadIdx.x & 31;
  const int m0 = wave * 16;

  f32x8 cir = {}, ciz = {}, cin = {}, chr_ = {}, chz = {}, chn = {};

  // gi tile: X[:, t_eff, :] @ W1^T  (K = 64)
  const __bf16* Ax = Xb + (size_t)m0 * (T * D) + (size_t)t_eff * D;
  const __bf16* W1 = W1b + (size_t)dir * (3 * H * D);
#pragma unroll
  for (int k0 = 0; k0 < D; k0 += 32) {
    bf16x16 a = frag_a(Ax, (size_t)T * D, lane, k0);
    cir = wmma(a, frag_b(W1, D, lane, j0, k0), cir);
    ciz = wmma(a, frag_b(W1 + (size_t)H * D, D, lane, j0, k0), ciz);
    cin = wmma(a, frag_b(W1 + (size_t)2 * H * D, D, lane, j0, k0), cin);
  }
  // gh tile: (gamma_h ⊙ h) @ Whh^T  (K = 256) -- hs_in already pre-scaled bf16
  const __bf16* Ah = hs_in + (size_t)dir * (B * H) + (size_t)m0 * H;
  const __bf16* Wh = Whhb + (size_t)dir * (3 * H * H);
#pragma unroll
  for (int k0 = 0; k0 < H; k0 += 32) {
    bf16x16 a = frag_a(Ah, H, lane, k0);
    chr_ = wmma(a, frag_b(Wh, H, lane, j0, k0), chr_);
    chz  = wmma(a, frag_b(Wh + (size_t)H * H, H, lane, j0, k0), chz);
    chn  = wmma(a, frag_b(Wh + (size_t)2 * H * H, H, lane, j0, k0), chn);
  }

  // GRU gate math directly on the C-tile layout (lane = col j, VGPR e = row)
  const int jl = lane & 15, hh = lane >> 4;
  const int j = j0 + jl;
  const float* cgd = cg + dir * 3 * H;
  const float* bhh = dir ? bhhb : bhhf;
  const float cg_r = cgd[j], cg_z = cgd[H + j], cg_n = cgd[2 * H + j];
  const float bh_r = bhh[j], bh_z = bhh[H + j], bh_n = bhh[2 * H + j];
  const float gam = gammah[dir * H + j];
  const float* hin = h_in + (size_t)dir * (B * H);
  float* hout = h_out + (size_t)dir * (B * H);
  __bf16* hsout = hs_out + (size_t)dir * (B * H);
#pragma unroll
  for (int e = 0; e < 8; ++e) {
    const int m = m0 + 8 * hh + e;
    const float hs = gam * hin[(size_t)m * H + j];  // f32 scaled state
    const float r = sigmoidf(cir[e] + cg_r + chr_[e] + bh_r);
    const float z = sigmoidf(ciz[e] + cg_z + chz[e] + bh_z);
    const float n = tanhf(cin[e] + cg_n + r * (chn[e] + bh_n));
    const float hn = (1.0f - z) * n + z * hs;
    hout[(size_t)m * H + j] = hn;
    hsout[(size_t)m * H + j] = (__bf16)(gam * hn);  // pre-scale for next step's A frags
    hsum[(size_t)m * (2 * H) + dir * H + j] += hn;  // pooling accumulator
  }
}

// ================= latent path (tiny, once) =================
__global__ void k_pool(const float* __restrict__ hsum, const float* __restrict__ tlW,
                       const float* __restrict__ tlb, float* __restrict__ z) {
  int i = blockIdx.x * blockDim.x + threadIdx.x;
  if (i < B * E) {
    int b = i / E, e = i % E;
    const float* hr = hsum + (size_t)b * 2 * H;
    const float* wr = tlW + (size_t)e * 2 * H;
    float s = 0.0f;
    for (int k = 0; k < 2 * H; ++k) s += hr[k] * wr[k];
    z[i] = s * (1.0f / T) + tlb[e];
  }
}
__global__ void k_seed(const float* __restrict__ z, const float* __restrict__ flW,
                       const float* __restrict__ flb, float* __restrict__ seed) {
  int i = blockIdx.x * blockDim.x + threadIdx.x;
  if (i < B * H) {
    int b = i / H, j = i % H;
    float s = flb[j];
    const float* zr = z + (size_t)b * E;
    const float* wr = flW + (size_t)j * E;
    for (int k = 0; k < E; ++k) s += zr[k] * wr[k];
    seed[i] = fmaxf(s, 0.0f);
  }
}
// step-invariant LSTM input gates, hoisted out of the decode loop
__global__ void k_gid(const float* __restrict__ seed, const float* __restrict__ Wih,
                      const float* __restrict__ bih, float* __restrict__ gid) {
  int i = blockIdx.x * blockDim.x + threadIdx.x;
  if (i < B * 4 * H) {
    int b = i / (4 * H), g = i % (4 * H);
    float s = bih[g];
    const float* sr = seed + (size_t)b * H;
    const float* wr = Wih + (size_t)g * H;
    for (int k = 0; k < H; ++k) s += sr[k] * wr[k];
    gid[i] = s;
  }
}

// ================= decoder LSTM step =================
__global__ void __launch_bounds__(256)
dec_step(const __bf16* __restrict__ Wb, const float* __restrict__ gid,
         const float* __restrict__ bhh, const __bf16* __restrict__ hs_in,
         __bf16* __restrict__ hs_out, float* __restrict__ c32,
         __bf16* __restrict__ hdec, int t) {
  const int j0 = blockIdx.x * 16;
  const int wave = threadIdx.x >> 5, lane = threadIdx.x & 31;
  const int m0 = wave * 16;
  f32x8 ci = {}, cf = {}, cgg = {}, co = {};
  const __bf16* Ah = hs_in + (size_t)m0 * H;
#pragma unroll
  for (int k0 = 0; k0 < H; k0 += 32) {
    bf16x16 a = frag_a(Ah, H, lane, k0);
    ci  = wmma(a, frag_b(Wb, H, lane, j0, k0), ci);
    cf  = wmma(a, frag_b(Wb + (size_t)H * H, H, lane, j0, k0), cf);
    cgg = wmma(a, frag_b(Wb + (size_t)2 * H * H, H, lane, j0, k0), cgg);
    co  = wmma(a, frag_b(Wb + (size_t)3 * H * H, H, lane, j0, k0), co);
  }
  const int jl = lane & 15, hh = lane >> 4;
  const int j = j0 + jl;
  const float b_i = bhh[j], b_f = bhh[H + j], b_g = bhh[2 * H + j], b_o = bhh[3 * H + j];
#pragma unroll
  for (int e = 0; e < 8; ++e) {
    const int m = m0 + 8 * hh + e;
    const float* g = gid + (size_t)m * 4 * H;
    const float gi_ = ci[e] + g[j] + b_i;
    const float gf  = cf[e] + g[H + j] + b_f;
    const float gg_ = cgg[e] + g[2 * H + j] + b_g;
    const float go  = co[e] + g[3 * H + j] + b_o;
    const float cold = c32[(size_t)m * H + j];
    const float cnew = sigmoidf(gf) * cold + sigmoidf(gi_) * tanhf(gg_);
    const float hnew = sigmoidf(go) * tanhf(cnew);
    c32[(size_t)m * H + j] = cnew;                      // same thread owns (m,j): in-place ok
    hs_out[(size_t)m * H + j] = (__bf16)hnew;           // ping-pong for next step's GEMM
    hdec[((size_t)m * T + t) * H + j] = (__bf16)hnew;   // row b*T+t for final projection
  }
}

// ================= final projection: (B*T,256) @ op_W^T -> (B*T,64) =================
__global__ void __launch_bounds__(256)
proj(const __bf16* __restrict__ hdec, const __bf16* __restrict__ opWb,
     const float* __restrict__ opb, float* __restrict__ out) {
  const int row0 = blockIdx.x * 64;
  const int wave = threadIdx.x >> 5, lane = threadIdx.x & 31;
  const int mt = wave & 3, n0 = (wave >> 2) * 32;
  f32x8 c0 = {}, c1 = {};
  const __bf16* Ab = hdec + (size_t)(row0 + mt * 16) * H;
#pragma unroll
  for (int k0 = 0; k0 < H; k0 += 32) {
    bf16x16 a = frag_a(Ab, H, lane, k0);
    c0 = wmma(a, frag_b(opWb, H, lane, n0, k0), c0);
    c1 = wmma(a, frag_b(opWb, H, lane, n0 + 16, k0), c1);
  }
  const int jl = lane & 15, hh = lane >> 4;
  const float b0 = opb[n0 + jl], b1 = opb[n0 + 16 + jl];
#pragma unroll
  for (int e = 0; e < 8; ++e) {
    const int r = row0 + mt * 16 + 8 * hh + e;
    out[(size_t)r * D + n0 + jl] = c0[e] + b0;
    out[(size_t)r * D + n0 + 16 + jl] = c1[e] + b1;
  }
}

// ================= host =================
extern "C" void kernel_launch(void* const* d_in, const int* in_sizes, int n_in,
                              void* d_out, int out_size, void* d_ws, size_t ws_size,
                              hipStream_t stream) {
  (void)in_sizes; (void)n_in; (void)out_size; (void)ws_size;
  const float* X        = (const float*)d_in[0];
  const float* f_Wih    = (const float*)d_in[1];
  const float* f_Whh    = (const float*)d_in[2];
  const float* f_bih    = (const float*)d_in[3];
  const float* f_bhh    = (const float*)d_in[4];
  const float* b_Wih    = (const float*)d_in[5];
  const float* b_Whh    = (const float*)d_in[6];
  const float* b_bih    = (const float*)d_in[7];
  const float* b_bhh    = (const float*)d_in[8];
  const float* Wdhf_b   = (const float*)d_in[10];
  const float* Wdhb_b   = (const float*)d_in[12];
  const float* tl_W     = (const float*)d_in[16];
  const float* tl_b     = (const float*)d_in[17];
  const float* fl_W     = (const float*)d_in[18];
  const float* fl_b     = (const float*)d_in[19];
  const float* lstm_Wih = (const float*)d_in[20];
  const float* lstm_Whh = (const float*)d_in[21];
  const float* lstm_bih = (const float*)d_in[22];
  const float* lstm_bhh = (const float*)d_in[23];
  const float* op_W     = (const float*)d_in[24];
  const float* op_b     = (const float*)d_in[25];
  float* out = (float*)d_out;

  char* w = (char*)d_ws;
  size_t o = 0;
  auto take = [&](size_t bytes) {
    char* p = w + o;
    o = (o + bytes + 255) & ~(size_t)255;
    return p;
  };
  __bf16* XB    = (__bf16*)take((size_t)B * T * D * 2);
  __bf16* W1B   = (__bf16*)take((size_t)2 * 3 * H * D * 2);
  __bf16* WHHB  = (__bf16*)take((size_t)2 * 3 * H * H * 2);
  __bf16* LSTMW = (__bf16*)take((size_t)4 * H * H * 2);
  __bf16* OPWB  = (__bf16*)take((size_t)D * H * 2);
  float*  CG    = (float*)take((size_t)2 * 3 * H * 4);
  float*  GAM   = (float*)take((size_t)2 * H * 4);
  float*  H32   = (float*)take((size_t)2 * 2 * B * H * 4);   // [buf][dir][B*H]
  __bf16* HSB   = (__bf16*)take((size_t)2 * 2 * B * H * 2);  // pre-scaled bf16
  float*  HSUM  = (float*)take((size_t)B * 2 * H * 4);
  float*  Z     = (float*)take((size_t)B * E * 4);
  float*  SEED  = (float*)take((size_t)B * H * 4);
  float*  GID   = (float*)take((size_t)B * 4 * H * 4);
  float*  C32   = (float*)take((size_t)B * H * 4);
  __bf16* DHS   = (__bf16*)take((size_t)2 * B * H * 2);      // decoder h ping-pong
  __bf16* HDEC  = (__bf16*)take((size_t)B * T * H * 2);

  // zero the recurrent state / accumulators (graph-capture-safe memsets)
  hipMemsetAsync(H32, 0, (size_t)2 * B * H * 4, stream);
  hipMemsetAsync(HSB, 0, (size_t)2 * B * H * 2, stream);
  hipMemsetAsync(HSUM, 0, (size_t)B * 2 * H * 4, stream);
  hipMemsetAsync(C32, 0, (size_t)B * H * 4, stream);
  hipMemsetAsync(DHS, 0, (size_t)B * H * 2, stream);

  const int thr = 256;
  k_f2b<<<2048, thr, 0, stream>>>(X, XB, B * T * D);
  k_f2b<<<192, thr, 0, stream>>>(f_Whh, WHHB, 3 * H * H);
  k_f2b<<<192, thr, 0, stream>>>(b_Whh, WHHB + (size_t)3 * H * H, 3 * H * H);
  k_f2b<<<256, thr, 0, stream>>>(lstm_Whh, LSTMW, 4 * H * H);
  k_f2b<<<64, thr, 0, stream>>>(op_W, OPWB, D * H);
  k_w1<<<(3 * H * D + thr - 1) / thr, thr, 0, stream>>>(f_Wih, W1B);
  k_w1<<<(3 * H * D + thr - 1) / thr, thr, 0, stream>>>(b_Wih, W1B + (size_t)3 * H * D);
  k_cg<<<(3 * H + thr - 1) / thr, thr, 0, stream>>>(f_Wih, f_bih, CG);
  k_cg<<<(3 * H + thr - 1) / thr, thr, 0, stream>>>(b_Wih, b_bih, CG + 3 * H);
  k_gamma<<<1, thr, 0, stream>>>(Wdhf_b, GAM, H);
  k_gamma<<<1, thr, 0, stream>>>(Wdhb_b, GAM + H, H);

  // encoder: 512 serial GRU-D steps, both directions in one launch (blockIdx.y)
  const size_t hstride = (size_t)2 * B * H;  // per ping-pong buffer
  for (int t = 0; t < T; ++t) {
    const int pi = t & 1;
    enc_step<<<dim3(16, 2), 256, 0, stream>>>(
        XB, W1B, WHHB, CG, GAM, f_bhh, b_bhh,
        H32 + (size_t)pi * hstride, HSB + (size_t)pi * hstride,
        H32 + (size_t)(1 - pi) * hstride, HSB + (size_t)(1 - pi) * hstride,
        HSUM, t);
  }

  k_pool<<<(B * E + thr - 1) / thr, thr, 0, stream>>>(HSUM, tl_W, tl_b, Z);
  k_seed<<<(B * H + thr - 1) / thr, thr, 0, stream>>>(Z, fl_W, fl_b, SEED);
  k_gid<<<(B * 4 * H + thr - 1) / thr, thr, 0, stream>>>(SEED, lstm_Wih, lstm_bih, GID);

  // decoder: 512 serial LSTM steps (input gates precomputed in GID)
  for (int t = 0; t < T; ++t) {
    const int pi = t & 1;
    dec_step<<<16, 256, 0, stream>>>(LSTMW, GID, lstm_bhh,
                                     DHS + (size_t)pi * B * H,
                                     DHS + (size_t)(1 - pi) * B * H,
                                     C32, HDEC, t);
  }

  // one big time-parallel projection: 65536 x 64 x 256
  proj<<<B * T / 64, 256, 0, stream>>>(HDEC, OPWB, op_b, out);
}
